// DeepGCN_slic_71081708748829
// MI455X (gfx1250) — matrix operand order repaired and checked by reading, not compiled, not run on hardware
//
#include <hip/hip_runtime.h>
#include <hip/hip_bf16.h>

// ---------------------------------------------------------------------------
// DeepGCN (ViG) forward for MI455X / gfx1250, wave32, WMMA f16->f32.
// v2: f16-preconverted weights (B frag = pure b128 loads, no VALU convert),
//     4 accumulators per wave (16x64 tile, A fragment amortized 4x),
//     guard-free epilogue for the exact-tile GEMMs.
// ---------------------------------------------------------------------------

#define BATCH 32
#define NTOK  196
#define CDIM  192
#define MTOT  (BATCH * NTOK)   // 6272 = 98*64
#define LBLK  12

typedef __attribute__((ext_vector_type(16))) _Float16 v16h;
typedef __attribute__((ext_vector_type(8)))  float    v8f;

__device__ __forceinline__ float gelu_f(float x) {
  // jax.nn.gelu default (approximate=True, tanh form)
  float x3 = x * x * x;
  return 0.5f * x * (1.0f + tanhf(0.7978845608028654f * (x + 0.044715f * x3)));
}

// ---------------------------------------------------------------------------
// Generic WMMA GEMM:  out[m][o] = epilogue( sum_k A[m][k] * W[o][k] )
// A: [M,K] row-major (fp32 or f16, template AT), W: [O,K] row-major f16.
// One wave computes a 16x64 tile (4 accumulators sharing one A fragment).
// Block = 4 waves covering 64 rows. Batched via blockIdx.z (element strides).
// Epilogue: v (+ pre[o]) (* g[o] + b[o]) (gelu) (+ res[m][o])
// GUARD=false requires M%64==0 (grid exact) and O%64==0.
// ---------------------------------------------------------------------------
template <typename AT, bool PRE, bool SCALE, bool GELU, bool RES, bool GUARD>
__global__ void __launch_bounds__(128)
wmma_gemm_kernel(const AT* __restrict__ A, const _Float16* __restrict__ W,
                 float* __restrict__ out, const float* __restrict__ res,
                 const float* __restrict__ pre, const float* __restrict__ g,
                 const float* __restrict__ bb,
                 int M, int K, int O,
                 long long sA, long long sW, long long sO) {
  const int lane = threadIdx.x & 31;
  const int wave = threadIdx.x >> 5;
  const int m0   = (blockIdx.x * 4 + wave) * 16;
  const int o0   = blockIdx.y * 64;
  const long long bz = (long long)blockIdx.z;
  A   += bz * sA;
  W   += bz * sW;
  out += bz * sO;
  const float* resp = res;
  if (RES) resp += bz * sO;
  if (GUARD && m0 >= M) return;     // wave-uniform: EXEC stays all-ones for WMMA

  const int half = lane >> 4;       // 0: lanes 0-15, 1: lanes 16-31
  const int l16  = lane & 15;
  int arow = m0 + l16;
  if (GUARD && arow >= M) arow = M - 1;     // clamp; OOB rows never stored
  const AT* Ap = A + (long long)arow * K;
  const _Float16* Wp[4];
#pragma unroll
  for (int t = 0; t < 4; ++t) {
    int wrow = o0 + t * 16 + l16;
    if (GUARD && wrow >= O) wrow = O - 1;   // clamp; OOB cols never stored
    Wp[t] = W + (long long)wrow * K;
  }
  const int akb = half * 8;         // A frag: K = akb..akb+7 and akb+16..akb+23
  const int bkb = half * 8;         // B frag: K = bkb..bkb+7 and bkb+16..bkb+23  (B[k][n] = W[o0+n][k])

  v8f acc[4];
#pragma unroll
  for (int t = 0; t < 4; ++t) acc[t] = (v8f){0.f,0.f,0.f,0.f,0.f,0.f,0.f,0.f};

  for (int kk = 0; kk < K; kk += 32) {
    v16h a;
#pragma unroll
    for (int e = 0; e < 8; ++e) {
      a[e]     = (_Float16)Ap[kk + akb + e];
      a[e + 8] = (_Float16)Ap[kk + akb + 16 + e];
    }
#pragma unroll
    for (int t = 0; t < 4; ++t) {
      v16h b;
#pragma unroll
      for (int e = 0; e < 16; ++e) b[e] = Wp[t][kk + bkb + (e < 8 ? e : e + 8)];
      acc[t] = __builtin_amdgcn_wmma_f32_16x16x32_f16(false, a, false, b,
                                                      (short)0, acc[t], false, false);
    }
  }

#pragma unroll
  for (int t = 0; t < 4; ++t) {
    const int col = o0 + t * 16 + l16;
    if (GUARD && col >= O) continue;
#pragma unroll
    for (int r = 0; r < 8; ++r) {
      const int row = m0 + half * 8 + r;   // D layout: VGPR r -> M = r (+8 hi lanes)
      if (GUARD && row >= M) continue;
      float v = acc[t][r];
      if (PRE)   v += pre[col];
      if (SCALE) v = v * g[col] + bb[col];
      if (GELU)  v = gelu_f(v);
      if (RES)   v += resp[(long long)row * O + col];
      out[(long long)row * O + col] = v;
    }
  }
}

// ---------------------------------------------------------------------------
// fp32 -> f16 weight preconversion
// ---------------------------------------------------------------------------
__global__ void cvt_f16_kernel(const float* __restrict__ src,
                               _Float16* __restrict__ dst, int n) {
  int i = blockIdx.x * blockDim.x + threadIdx.x;
  if (i < n) dst[i] = (_Float16)src[i];
}

// ---------------------------------------------------------------------------
// x [B,11,14,14] -> xt [B*N, 11] tokens (h = reshape(b,11,196).transpose)
// ---------------------------------------------------------------------------
__global__ void xpose_kernel(const float* __restrict__ x, float* __restrict__ xt) {
  int i = blockIdx.x * blockDim.x + threadIdx.x;
  if (i >= MTOT * 11) return;
  int m = i / 11, c = i % 11;
  int b = m / NTOK, n = m % NTOK;
  xt[i] = x[((long long)b * 11 + c) * NTOK + n];
}

// ---------------------------------------------------------------------------
// Naive pointwise conv + BN (+gelu) for the small/odd stem dims.
// ---------------------------------------------------------------------------
template <bool GELU>
__global__ void stem_pw_kernel(const float* __restrict__ A, const float* __restrict__ W,
                               const float* __restrict__ g, const float* __restrict__ bb,
                               float* __restrict__ out, int M, int K, int O) {
  int i = blockIdx.x * blockDim.x + threadIdx.x;
  if (i >= M * O) return;
  int m = i / O, o = i % O;
  const float* ar = A + (long long)m * K;
  const float* wr = W + (long long)o * K;
  float s = 0.f;
  for (int k = 0; k < K; ++k) s += ar[k] * wr[k];
  s = s * g[o] + bb[o];
  if (GELU) s = gelu_f(s);
  out[i] = s;
}

// ---------------------------------------------------------------------------
// Per token: S = sum(y^2); sq = S/(S+eps); yn_h = f16(y * rsqrt(S+eps))
// yn is consumed only by the f16 score GEMM, so emit it in f16 directly.
// ---------------------------------------------------------------------------
__global__ void norm_sq_kernel(const float* __restrict__ y,
                               _Float16* __restrict__ yn_h, float* __restrict__ sq) {
  int m = blockIdx.x * blockDim.x + threadIdx.x;
  if (m >= MTOT) return;
  const float* yr = y + (long long)m * CDIM;
  float s = 0.f;
  for (int c = 0; c < CDIM; ++c) { float v = yr[c]; s += v * v; }
  float d  = s + 1e-12f;
  sq[m]    = s / d;                       // ||yn||^2
  float rs = rsqrtf(d);
  _Float16* o = yn_h + (long long)m * CDIM;
  for (int c = 0; c < CDIM; ++c) o[c] = (_Float16)(yr[c] * rs);
}

// ---------------------------------------------------------------------------
// One wave per token: scores = 2*dot - sq_n - sq_m, iterative top-k argmax
// (wave32 shfl_xor reduction, ties -> lowest index, matching lax.top_k),
// fused max-relative aggregation: msg[c] = max_t y[idx_t][c] - y[n][c].
// Writes zcat = concat(y, msg)  [M, 2C].  No LDS, no barriers.
// ---------------------------------------------------------------------------
__global__ void __launch_bounds__(256)
knn_msg_kernel(const float* __restrict__ dot, const float* __restrict__ sq,
               const float* __restrict__ y, float* __restrict__ zcat, int k) {
  const int lane = threadIdx.x & 31;
  const int wtok = blockIdx.x * (blockDim.x >> 5) + (threadIdx.x >> 5);
  if (wtok >= MTOT) return;               // exact grid: never taken, but safe
  const int b = wtok / NTOK;
  const float* drow = dot + ((long long)wtok) * NTOK;   // dot[b][n][*]
  const float  sqn  = sq[wtok];
  const float* sqb  = sq + (long long)b * NTOK;

  float sc[7];
#pragma unroll
  for (int j = 0; j < 7; ++j) {
    int m2 = lane + 32 * j;
    sc[j] = (m2 < NTOK) ? (2.f * drow[m2] - sqn - sqb[m2]) : -1e30f;
  }
  float mx[6];
#pragma unroll
  for (int jc = 0; jc < 6; ++jc) mx[jc] = -1e30f;

  const float* ybase = y + (long long)b * NTOK * CDIM;
  for (int t = 0; t < k; ++t) {
    float bv = -1e30f; int bi = 0;
#pragma unroll
    for (int j = 0; j < 7; ++j) {
      int m2 = lane + 32 * j;
      if (sc[j] > bv) { bv = sc[j]; bi = m2; }
    }
    for (int off = 16; off >= 1; off >>= 1) {   // wave32 argmax, stable ties
      float ov = __shfl_xor(bv, off, 32);
      int   oi = __shfl_xor(bi, off, 32);
      if (ov > bv || (ov == bv && oi < bi)) { bv = ov; bi = oi; }
    }
    // bi is now wave-uniform: knock it out and accumulate its features
#pragma unroll
    for (int j = 0; j < 7; ++j)
      if ((bi >> 5) == j && (bi & 31) == lane) sc[j] = -1e30f;
    const float* yr = ybase + (long long)bi * CDIM;
#pragma unroll
    for (int jc = 0; jc < 6; ++jc)
      mx[jc] = fmaxf(mx[jc], yr[lane + 32 * jc]);
  }

  const float* yn0 = y + (long long)wtok * CDIM;
  float* zr = zcat + (long long)wtok * (2 * CDIM);
#pragma unroll
  for (int jc = 0; jc < 6; ++jc) {
    int c = lane + 32 * jc;
    float yv = yn0[c];
    zr[c]        = yv;
    zr[CDIM + c] = mx[jc] - yv;
  }
}

// ---------------------------------------------------------------------------
// Mean over tokens: pooled[b][c] = mean_n h[b][n][c]
// ---------------------------------------------------------------------------
__global__ void pool_kernel(const float* __restrict__ h, float* __restrict__ pooled) {
  int i = blockIdx.x * blockDim.x + threadIdx.x;
  if (i >= BATCH * CDIM) return;
  int b = i / CDIM, c = i % CDIM;
  float s = 0.f;
  for (int n = 0; n < NTOK; ++n) s += h[((long long)b * NTOK + n) * CDIM + c];
  pooled[i] = s * (1.f / (float)NTOK);
}

// ---------------------------------------------------------------------------
// Host side
// ---------------------------------------------------------------------------
static inline dim3 gemm_grid(int M, int O, int batches) {
  return dim3((unsigned)((M + 63) / 64), (unsigned)((O + 63) / 64), (unsigned)batches);
}

static inline void cvt_w(hipStream_t s, const float* src, _Float16* dst, int n) {
  cvt_f16_kernel<<<(n + 255) / 256, 256, 0, s>>>(src, dst, n);
}

extern "C" void kernel_launch(void* const* d_in, const int* in_sizes, int n_in,
                              void* d_out, int out_size, void* d_ws, size_t ws_size,
                              hipStream_t stream) {
  (void)in_sizes; (void)n_in; (void)out_size; (void)ws_size;
  const float* x = (const float*)d_in[0];
  const float *sw[5], *sg[5], *sb[5];
  for (int i = 0; i < 5; ++i) {
    sw[i] = (const float*)d_in[1 + 3 * i];
    sg[i] = (const float*)d_in[2 + 3 * i];
    sb[i] = (const float*)d_in[3 + 3 * i];
  }
  const float* fc1_w  = (const float*)d_in[16];
  const float* fc1_g  = (const float*)d_in[17];
  const float* fc1_b  = (const float*)d_in[18];
  const float* gc_w   = (const float*)d_in[19];
  const float* gc_g   = (const float*)d_in[20];
  const float* gc_b   = (const float*)d_in[21];
  const float* fc2_w  = (const float*)d_in[22];
  const float* fc2_g  = (const float*)d_in[23];
  const float* fc2_b  = (const float*)d_in[24];
  const float* ffn1_w = (const float*)d_in[25];
  const float* ffn1_g = (const float*)d_in[26];
  const float* ffn1_b = (const float*)d_in[27];
  const float* ffn2_w = (const float*)d_in[28];
  const float* ffn2_g = (const float*)d_in[29];
  const float* ffn2_b = (const float*)d_in[30];
  const float* pw1    = (const float*)d_in[31];
  const float* pb1    = (const float*)d_in[32];
  const float* pg1    = (const float*)d_in[33];
  const float* pgb1   = (const float*)d_in[34];
  const float* pw2    = (const float*)d_in[35];
  const float* pb2    = (const float*)d_in[36];

  // --- workspace layout ---
  // float region (~38.8 MB), then f16 region (~15.6 MB). Total < 55 MB.
  float* wsf  = (float*)d_ws;
  float* h    = wsf;                            // [M, C]
  float* y    = h    + (size_t)MTOT * CDIM;     // [M, C]
  float* zcat = y    + (size_t)MTOT * CDIM;     // [M, 2C]
  float* ffn  = zcat + (size_t)MTOT * 2 * CDIM; // [M, 4C]
  float* sq   = ffn  + (size_t)MTOT * 4 * CDIM; // [M]
  float* pooled = sq + MTOT;                    // [B, C]
  float* p1   = pooled + BATCH * CDIM;          // [B, 1024]
  // Aliases (disjoint in time):
  float* dotb = ffn;                            // [B, N, N] before ffn1 runs
  float* xt   = zcat;                           // [M, 11] stem input

  _Float16* wsh   = (_Float16*)(p1 + BATCH * 1024);
  const int nC2   = CDIM * CDIM;                // 36864
  _Float16* wh_fc1  = wsh;                            // [12, C, C]
  _Float16* wh_gc   = wh_fc1  + (size_t)LBLK * nC2;   // [12, C, 2C]
  _Float16* wh_fc2  = wh_gc   + (size_t)LBLK * 2 * nC2;
  _Float16* wh_ffn1 = wh_fc2  + (size_t)LBLK * nC2;   // [12, 4C, C]
  _Float16* wh_ffn2 = wh_ffn1 + (size_t)LBLK * 4 * nC2;
  _Float16* wh_pw1  = wh_ffn2 + (size_t)LBLK * 4 * nC2; // [1024, C]
  _Float16* wh_pw2  = wh_pw1  + (size_t)1024 * CDIM;    // [1000, 1024]
  _Float16* yn_h    = wh_pw2  + (size_t)1000 * 1024;    // [M, C]

  const int knum[LBLK] = {9, 9, 10, 11, 12, 13, 13, 14, 15, 16, 17, 18};

  // ---- weight preconversion (cheap; once per launch) ----
  cvt_w(stream, fc1_w,  wh_fc1,  LBLK * nC2);
  cvt_w(stream, gc_w,   wh_gc,   LBLK * 2 * nC2);
  cvt_w(stream, fc2_w,  wh_fc2,  LBLK * nC2);
  cvt_w(stream, ffn1_w, wh_ffn1, LBLK * 4 * nC2);
  cvt_w(stream, ffn2_w, wh_ffn2, LBLK * 4 * nC2);
  cvt_w(stream, pw1,    wh_pw1,  1024 * CDIM);
  cvt_w(stream, pw2,    wh_pw2,  1000 * 1024);

  // ---- stem ----
  {
    int n = MTOT * 11;
    xpose_kernel<<<(n + 255) / 256, 256, 0, stream>>>(x, xt);
    const int dims[6] = {11, 24, 48, 96, 192, 192};
    const float* src = xt;
    float* pp[5] = {y, ffn, y, ffn, h};   // ping-pong scratch, final -> h
    for (int i = 0; i < 5; ++i) {
      int M = MTOT, K = dims[i], O = dims[i + 1];
      int nt = M * O;
      if (i < 4)
        stem_pw_kernel<true ><<<(nt + 255) / 256, 256, 0, stream>>>(src, sw[i], sg[i], sb[i], pp[i], M, K, O);
      else
        stem_pw_kernel<false><<<(nt + 255) / 256, 256, 0, stream>>>(src, sw[i], sg[i], sb[i], pp[i], M, K, O);
      src = pp[i];
    }
  }

  // ---- 12 Grapher + FFN blocks ----
  for (int i = 0; i < LBLK; ++i) {
    // fc1: y = h @ fc1_w^T * g + b
    wmma_gemm_kernel<float, false, true, false, false, false>
        <<<gemm_grid(MTOT, CDIM, 1), 128, 0, stream>>>(
        h, wh_fc1 + (size_t)i * nC2, y, nullptr, nullptr,
        fc1_g + i * CDIM, fc1_b + i * CDIM, MTOT, CDIM, CDIM, 0, 0, 0);
    // normalize + squared-norm terms (yn emitted in f16 for the score GEMM)
    norm_sq_kernel<<<(MTOT + 255) / 256, 256, 0, stream>>>(y, yn_h, sq);
    // batched score GEMM: dot[b] = yn_b @ yn_b^T   (196x196x192, pure f16 in)
    wmma_gemm_kernel<_Float16, false, false, false, false, true>
        <<<gemm_grid(NTOK, NTOK, BATCH), 128, 0, stream>>>(
        yn_h, yn_h, dotb, nullptr, nullptr, nullptr, nullptr,
        NTOK, CDIM, NTOK,
        (long long)NTOK * CDIM, (long long)NTOK * CDIM, (long long)NTOK * NTOK);
    // top-k + max-relative message -> zcat = concat(y, msg)
    knn_msg_kernel<<<MTOT / 8, 256, 0, stream>>>(dotb, sq, y, zcat, knum[i]);
    // graph conv: zg = gelu(zcat @ gc_w^T * g + b)   (reuse y)
    wmma_gemm_kernel<float, false, true, true, false, false>
        <<<gemm_grid(MTOT, CDIM, 1), 128, 0, stream>>>(
        zcat, wh_gc + (size_t)i * 2 * nC2, y, nullptr, nullptr,
        gc_g + i * CDIM, gc_b + i * CDIM, MTOT, 2 * CDIM, CDIM, 0, 0, 0);
    // fc2 + residual: h = zg @ fc2_w^T * g + b + h
    wmma_gemm_kernel<float, false, true, false, true, false>
        <<<gemm_grid(MTOT, CDIM, 1), 128, 0, stream>>>(
        y, wh_fc2 + (size_t)i * nC2, h, h, nullptr,
        fc2_g + i * CDIM, fc2_b + i * CDIM, MTOT, CDIM, CDIM, 0, 0, 0);
    // ffn1: f = gelu(h @ ffn1_w^T * g + b)
    wmma_gemm_kernel<float, false, true, true, false, false>
        <<<gemm_grid(MTOT, 4 * CDIM, 1), 128, 0, stream>>>(
        h, wh_ffn1 + (size_t)i * 4 * nC2, ffn, nullptr, nullptr,
        ffn1_g + i * 4 * CDIM, ffn1_b + i * 4 * CDIM, MTOT, CDIM, 4 * CDIM, 0, 0, 0);
    // ffn2 + residual: h = f @ ffn2_w^T * g + b + h
    wmma_gemm_kernel<float, false, true, false, true, false>
        <<<gemm_grid(MTOT, CDIM, 1), 128, 0, stream>>>(
        ffn, wh_ffn2 + (size_t)i * 4 * nC2, h, h, nullptr,
        ffn2_g + i * CDIM, ffn2_b + i * CDIM, MTOT, 4 * CDIM, CDIM, 0, 0, 0);
  }

  // ---- head ----
  pool_kernel<<<(BATCH * CDIM + 255) / 256, 256, 0, stream>>>(h, pooled);
  // p1 = gelu((pooled @ pw1^T + pb1) * pg1 + pgb1)
  wmma_gemm_kernel<float, true, true, true, false, true>
      <<<gemm_grid(BATCH, 1024, 1), 128, 0, stream>>>(
      pooled, wh_pw1, p1, nullptr, pb1, pg1, pgb1, BATCH, CDIM, 1024, 0, 0, 0);
  // out = p1 @ pw2^T + pb2
  wmma_gemm_kernel<float, true, false, false, false, true>
      <<<gemm_grid(BATCH, 1000, 1), 128, 0, stream>>>(
      p1, wh_pw2, (float*)d_out, nullptr, pb2, nullptr, nullptr, BATCH, 1024, 1000, 0, 0, 0);
}
